// ScaledDotProductAttention_67740224193337
// MI455X (gfx1250) — matrix operand order repaired and checked by reading, not compiled, not run on hardware
//
#include <hip/hip_runtime.h>

typedef __attribute__((ext_vector_type(16))) _Float16 v16h;
typedef __attribute__((ext_vector_type(8)))  _Float16 v8h;
typedef __attribute__((ext_vector_type(4)))  _Float16 v4h;
typedef __attribute__((ext_vector_type(8)))  float    v8f;
typedef __attribute__((ext_vector_type(4)))  unsigned int v4u;
typedef __attribute__((ext_vector_type(8)))  int      v8i;
typedef __attribute__((ext_vector_type(4)))  int      v4i;

#define S_LEN 2048
#define HD    64
#define BHEAD 32
#define NBAT  2
#define BR    64      // query rows per workgroup (16 per wave)
#define BC    64      // keys per iteration
#define NWAVE 4
// log2(e) folded into Q scaling and the mask constant: weights = exp2(score*log2e)
#define QSCALE   (0.125f * 1.4426950408889634f)
#define MASK_L2  (-6.1960912e9f)          // (-2^32+1) * log2(e)

// ---------------- pre-pass: f32 -> f16, V transposed per head ----------------
__global__ __launch_bounds__(256)
void cvt_kernel(const float* __restrict__ K, const float* __restrict__ V,
                _Float16* __restrict__ K16, _Float16* __restrict__ VT16)
{
    int idx4 = blockIdx.x * 256 + threadIdx.x;       // float4 index
    const float4 kf = ((const float4*)K)[idx4];
    const float4 vf = ((const float4*)V)[idx4];
    int idx = idx4 * 4;
    v4h kh; kh[0] = (_Float16)kf.x; kh[1] = (_Float16)kf.y;
            kh[2] = (_Float16)kf.z; kh[3] = (_Float16)kf.w;
    *(v4h*)&K16[idx] = kh;                            // same layout [bh][key][d]
    int bh  = idx >> 17;                              // / (S*HD)
    int rem = idx & (S_LEN * HD - 1);
    int key = rem >> 6;
    int d   = rem & 63;                               // d..d+3 contiguous
    _Float16* vt = VT16 + ((size_t)bh << 17) + key;   // [bh][d][key]
    vt[(size_t)(d + 0) * S_LEN] = (_Float16)vf.x;
    vt[(size_t)(d + 1) * S_LEN] = (_Float16)vf.y;
    vt[(size_t)(d + 2) * S_LEN] = (_Float16)vf.z;
    vt[(size_t)(d + 3) * S_LEN] = (_Float16)vf.w;
}

// ---- issue one 2D TDM tile load: global (f16, row stride0) -> LDS row-major ----
__device__ __forceinline__ void tdm_load_2d(unsigned lds_off, unsigned long long gaddr,
                                            unsigned tile_d0, unsigned tile_d1,
                                            unsigned long long stride0)
{
    v4u g0;
    g0.x = 1u;                                        // count=1, user descriptor
    g0.y = lds_off;                                   // LDS byte address
    g0.z = (unsigned)gaddr;                           // global_addr[31:0]
    g0.w = (unsigned)((gaddr >> 32) & 0x01FFFFFFull)  // global_addr[56:32]
         | (2u << 30);                                // type = 2 ("image")
    v8i g1;
    g1[0] = (int)(1u << 16);                          // wg_mask=0, data_size=1 (2B)
    g1[1] = (int)((tile_d0 & 0xFFFFu) << 16);         // tensor_dim0 = tile_d0
    g1[2] = (int)((tile_d1 & 0xFFFFu) << 16);         // tensor_dim0 hi=0, tensor_dim1 = tile_d1
    g1[3] = (int)((tile_d0 & 0xFFFFu) << 16);         // tensor_dim1 hi=0, tile_dim0
    g1[4] = (int)(tile_d1 & 0xFFFFu);                 // tile_dim1, tile_dim2=0
    g1[5] = (int)(unsigned)stride0;                   // tensor_dim0_stride[31:0]
    g1[6] = (int)((stride0 >> 32) & 0xFFFFull);       // stride0[47:32], stride1 lo=0
    g1[7] = 0;
    v4i z4 = {0, 0, 0, 0};
    v8i z8 = {0, 0, 0, 0, 0, 0, 0, 0};
    // clang-23 / therock-10.0 lane: 6-arg form (g0, g1, g2, g3, g4, cpol)
    __builtin_amdgcn_tensor_load_to_lds(g0, g1, z4, z4, z8, 0);
}

// ------------------------------- attention -------------------------------
__global__ __launch_bounds__(128)
void fattn_wmma_kernel(const float* __restrict__ Q,
                       const int*   __restrict__ masks,
                       const _Float16* __restrict__ K16,
                       const _Float16* __restrict__ VT16,
                       float* __restrict__ Out)
{
    __shared__ _Float16 klds[2][BC * HD];             // [key][d]  (2 x 8 KB)
    __shared__ _Float16 vlds[2][HD * BC];             // [d][key]  (2 x 8 KB)
    __shared__ _Float16 plds[NWAVE][16 * BC];         // per-wave P tile (8 KB)

    const int tid  = threadIdx.x;
    const int lane = tid & 31;
    const int wave = tid >> 5;
    const int h    = lane >> 4;
    const int ln   = lane & 15;
    const int bh    = blockIdx.y;
    const int qtile = blockIdx.x;
    const int batch = bh & (NBAT - 1);                // jnp.tile -> row i uses masks[i%2]

    const float*    qp  = Q    + (size_t)bh * S_LEN * HD;
    const _Float16* kp  = K16  + (size_t)bh * S_LEN * HD;   // [key][d]
    const _Float16* vtp = VT16 + (size_t)bh * S_LEN * HD;   // [d][key]
    float*          op  = Out  + (size_t)bh * S_LEN * HD;
    const int*      mp  = masks + batch * S_LEN;

    const int qrow0 = qtile * BR + wave * 16;

    // Q A-fragments (contiguous k order, same as K B-fragments): K(dim) = j + 16*(lane>=16)
    v16h aq0, aq1;
    {
        const float* qr = qp + (size_t)(qrow0 + ln) * HD;
        #pragma unroll
        for (int j = 0; j < 16; ++j) {
            int d0 = j + (h ? 16 : 0);
            aq0[j] = (_Float16)(qr[d0]      * QSCALE);
            aq1[j] = (_Float16)(qr[d0 + 32] * QSCALE);
        }
    }

    v8f o[4] = {v8f{}, v8f{}, v8f{}, v8f{}};
    float lsum[8];
    #pragma unroll
    for (int r = 0; r < 8; ++r) lsum[r] = 0.0f;

    _Float16* pw = &plds[wave][0];
    const unsigned kloff0 = (unsigned)(size_t)&klds[0][0];
    const unsigned kloff1 = (unsigned)(size_t)&klds[1][0];
    const unsigned vloff0 = (unsigned)(size_t)&vlds[0][0];
    const unsigned vloff1 = (unsigned)(size_t)&vlds[1][0];

    const int NIT = S_LEN / BC;                       // 32
    // prologue: wave 0 kicks off tile 0 on the Tensor Data Mover
    if (wave == 0) {
        tdm_load_2d(kloff0, (unsigned long long)(size_t)kp,  HD, BC, HD);
        tdm_load_2d(vloff0, (unsigned long long)(size_t)vtp, BC, HD, S_LEN);
    }

    for (int kb = 0; kb < NIT; ++kb) {
        const int c0  = kb * BC;
        const int buf = kb & 1;

        if (wave == 0)
            __builtin_amdgcn_s_wait_tensorcnt((short)0);   // tile kb landed in LDS
        __syncthreads();                                   // visible to all waves;
                                                           // also: compute kb-1 done
        if (wave == 0 && kb + 1 < NIT) {
            const int c1 = c0 + BC, b1 = (kb + 1) & 1;
            tdm_load_2d(b1 ? kloff1 : kloff0,
                        (unsigned long long)(size_t)(kp + (size_t)c1 * HD), HD, BC, HD);
            tdm_load_2d(b1 ? vloff1 : vloff0,
                        (unsigned long long)(size_t)(vtp + c1), BC, HD, S_LEN);
        }

        const _Float16* kl = &klds[buf][0];
        const _Float16* vl = &vlds[buf][0];

        // ---- S = Q*K^T : four 16-key tiles, K-dim split 0..31 / 32..63
        v8f s[4];
        #pragma unroll
        for (int t = 0; t < 4; ++t) {
            v16h b0 = *(const v16h*)&kl[(t * 16 + ln) * HD + 16 * h];
            v16h b1 = *(const v16h*)&kl[(t * 16 + ln) * HD + 16 * h + 32];
            v8f acc = {};
            acc = __builtin_amdgcn_wmma_f32_16x16x32_f16(false, aq0, false, b0, (short)0, acc, false, false);
            acc = __builtin_amdgcn_wmma_f32_16x16x32_f16(false, aq1, false, b1, (short)0, acc, false, false);
            s[t] = acc;
        }

        // ---- mask + exp2 (scores pre-scaled by log2e; no max-subtraction needed)
        #pragma unroll
        for (int t = 0; t < 4; ++t) {
            float madd = mp[c0 + t * 16 + ln] ? MASK_L2 : 0.0f;
            #pragma unroll
            for (int r = 0; r < 8; ++r) {
                float e = __builtin_amdgcn_exp2f(s[t][r] + madd);
                lsum[r] += e;
                pw[(r + 8 * h) * BC + t * 16 + ln] = (_Float16)e;
            }
        }
        asm volatile("s_wait_dscnt 0" ::: "memory");       // own-wave LDS RAW

        // ---- rebuild P as two A-fragments (keys 0..31, 32..63)
        v16h ap0, ap1;
        {
            v8h l0 = *(const v8h*)&pw[ln * BC + 16 * h];
            v8h h0 = *(const v8h*)&pw[ln * BC + 16 * h + 8];
            v8h l1 = *(const v8h*)&pw[ln * BC + 16 * h + 32];
            v8h h1 = *(const v8h*)&pw[ln * BC + 16 * h + 40];
            #pragma unroll
            for (int j = 0; j < 8; ++j) {
                ap0[j] = l0[j]; ap0[j + 8] = h0[j];
                ap1[j] = l1[j]; ap1[j + 8] = h1[j];
            }
        }

        // ---- O += P * V : 2 key halves x 4 dim blocks
        #pragma unroll
        for (int nb = 0; nb < 4; ++nb) {
            v16h bv0 = *(const v16h*)&vl[(nb * 16 + ln) * BC + 16 * h];
            v16h bv1 = *(const v16h*)&vl[(nb * 16 + ln) * BC + 32 + 16 * h];
            o[nb] = __builtin_amdgcn_wmma_f32_16x16x32_f16(false, ap0, false, bv0, (short)0, o[nb], false, false);
            o[nb] = __builtin_amdgcn_wmma_f32_16x16x32_f16(false, ap1, false, bv1, (short)0, o[nb], false, false);
        }
    }

    // ---- one deferred row-sum reduction over the 16-lane half, then O / l
    #pragma unroll
    for (int r = 0; r < 8; ++r) {
        float rs = lsum[r];
        #pragma unroll
        for (int off = 1; off < 16; off <<= 1)
            rs += __shfl_xor(rs, off, 32);
        float inv = 1.0f / rs;
        float* orow = op + (size_t)(qrow0 + r + 8 * h) * HD + ln;
        orow[0]  = o[0][r] * inv;
        orow[16] = o[1][r] * inv;
        orow[32] = o[2][r] * inv;
        orow[48] = o[3][r] * inv;
    }
}

extern "C" void kernel_launch(void* const* d_in, const int* in_sizes, int n_in,
                              void* d_out, int out_size, void* d_ws, size_t ws_size,
                              hipStream_t stream) {
    const float* q     = (const float*)d_in[0];
    const float* k     = (const float*)d_in[1];
    const float* v     = (const float*)d_in[2];
    const int*   masks = (const int*)d_in[3];
    float*       out   = (float*)d_out;
    (void)in_sizes; (void)n_in; (void)out_size; (void)ws_size;

    _Float16* k16  = (_Float16*)d_ws;                       // 8 MB
    _Float16* vt16 = k16 + (size_t)BHEAD * S_LEN * HD;      // 8 MB

    int nElem4 = BHEAD * S_LEN * HD / 4;
    cvt_kernel<<<nElem4 / 256, 256, 0, stream>>>(k, v, k16, vt16);

    dim3 grid(S_LEN / BR, BHEAD);                           // 32 x 32 workgroups
    dim3 block(128);                                        // 4 waves of 32
    fattn_wmma_kernel<<<grid, block, 0, stream>>>(q, masks, k16, vt16, out);
}
